// S4Model_52853867544646
// MI455X (gfx1250) — matrix operand order repaired and checked by reading, not compiled, not run on hardware
//
#include <hip/hip_runtime.h>
#include <hip/hip_bf16.h>
#include <math.h>

// ---------------------------------------------------------------------------
// S4D forward on gfx1250 (MI455X): chunked state-space conv via bf16 WMMA.
// - Weights / x pre-packed into WMMA fragment order (32B per lane per frag).
// - Conv kernels stage the shared u-chunk (and s0) in LDS once per block.
// - GEMMs: 2x4 register blocking (no spills) + block-level B staged in LDS.
// ---------------------------------------------------------------------------

typedef __attribute__((ext_vector_type(16))) __bf16 v16bf;
typedef __attribute__((ext_vector_type(8)))  float  v8f;

#define D_INPUT  100
#define D_MODEL  256
#define D_OUTPUT 100
#define N2       32
#define B_SZ     16
#define L_SEQ    4096
#define T_CHUNK  64
#define N_CHUNK  (L_SEQ / T_CHUNK)   // 64
#define M_ROWS   (B_SZ * L_SEQ)      // 65536
#define D_HID    (D_MODEL * 2)       // 512
#define LDS_PITCH 68                 // 64 + 4 pad floats: spreads LDS banks

static __device__ __forceinline__ __bf16 f2bf(float f) { return (__bf16)f; }
static __device__ __forceinline__ float bf2f(__bf16 h) { return (float)h; }
static __device__ __forceinline__ float gelu_f(float x) {
  return 0.5f * x * (1.0f + erff(x * 0.7071067811865476f));
}
static __device__ __forceinline__ v8f wmma_bf16(v16bf a, v16bf b, v8f c) {
  // (neg_a, A, neg_b, B, c_mod, C, reuse_a, reuse_b)
  return __builtin_amdgcn_wmma_f32_16x16x32_bf16(false, a, false, b, (short)0, c,
                                                 false, false);
}

// A-fragment element order (16-bit A 16x32, wave32):
//   lane = (m & 15) | (khalf << 4);  a[i] = src[k0 + 8*khalf + i + (i>=8 ? 8 : 0)]
// B-fragment element order (16-bit B 32x16, wave32):
//   lane = (n & 15) | (khalf << 4);  b[e] = B[k0 + 8*khalf + e + (e>=8 ? 8 : 0), n]

// ---------------------------------------------------------------------------
// 0a) Pack row-major weight W[Nrows x K] (f32) into B fragments (bf16).
// ---------------------------------------------------------------------------
__global__ void pack_b_kernel(const float* __restrict__ W, __bf16* __restrict__ P,
                              int Nrows, int K, int KS) {
  int tid = blockIdx.x * blockDim.x + threadIdx.x;
  int lane = tid & 31;
  int frag = tid >> 5;            // nt*KS + ks
  int ks = frag % KS;
  int nt = frag / KS;
  int n = nt * 16 + (lane & 15);
  int k0 = ks * 32 + ((lane >> 4) << 3);
  __bf16* dst = P + (size_t)tid * 16;
  const float* src = W + (size_t)n * K;
  bool nok = n < Nrows;
#pragma unroll
  for (int e = 0; e < 16; ++e) {
    int k = k0 + e + (e >= 8 ? 8 : 0);
    dst[e] = f2bf((nok && k < K) ? src[k] : 0.0f);
  }
}

// ---------------------------------------------------------------------------
// 0b) Pack x[M x 100] (f32) into A fragments (bf16), K padded to 128.
// ---------------------------------------------------------------------------
__global__ void pack_a_x_kernel(const float* __restrict__ X, __bf16* __restrict__ P) {
  int tid = blockIdx.x * blockDim.x + threadIdx.x;   // mtiles*4*32
  int lane = tid & 31;
  int frag = tid >> 5;
  int ks = frag & 3;
  int mt = frag >> 2;
  int m = mt * 16 + (lane & 15);
  int kb = ks * 32 + 8 * (lane >> 4);
  const float* src = X + (size_t)m * D_INPUT;
  __bf16* dst = P + (size_t)tid * 16;
#pragma unroll
  for (int e = 0; e < 16; ++e) {
    int k = kb + e + (e >= 8 ? 8 : 0);
    dst[e] = f2bf(k < D_INPUT ? src[k] : 0.0f);
  }
}

// ---------------------------------------------------------------------------
// 1) Per-channel tables: Toeplitz kernel K_toep (T x T), initial-state
//    projector P (T x 2N), end-state collector Q (2N x T), propagator w^T.
// ---------------------------------------------------------------------------
__global__ void s4_tables_kernel(const float* __restrict__ log_dt,
                                 const float* __restrict__ log_A_real,
                                 const float* __restrict__ A_imag,
                                 const float* __restrict__ B_re,
                                 const float* __restrict__ B_im,
                                 const float* __restrict__ C_re,
                                 const float* __restrict__ C_im,
                                 __bf16* __restrict__ Ktoep,
                                 __bf16* __restrict__ Ptab,
                                 __bf16* __restrict__ Qtab,
                                 float* __restrict__ wT_re,
                                 float* __restrict__ wT_im) {
  __shared__ float s_ar[N2], s_ai[N2], s_cr[N2], s_ci[N2], s_k[T_CHUNK];
  int h = blockIdx.x;
  int t = threadIdx.x;                     // 0..63
  if (t < N2) {
    int n = t;
    float dt = expf(log_dt[h]);
    float Ar = -expf(log_A_real[h * N2 + n]);
    float Ai = A_imag[h * N2 + n];
    float br = B_re[h * N2 + n], bi = B_im[h * N2 + n];
    float cr = C_re[h * N2 + n], ci = C_im[h * N2 + n];
    float Cr = br * cr - bi * ci, Ci = br * ci + bi * cr;  // fold B into C
    float ar = dt * Ar, ai = dt * Ai;                      // dtA
    float e = expf(ar);
    float wr = e * cosf(ai), wi = e * sinf(ai);            // w = exp(dtA)
    float numr = wr - 1.0f, numi = wi;                     // ZOH: C' = C(w-1)/A
    float den = Ar * Ar + Ai * Ai;
    float qr = (numr * Ar + numi * Ai) / den;
    float qi = (numi * Ar - numr * Ai) / den;
    s_cr[n] = Cr * qr - Ci * qi;
    s_ci[n] = Cr * qi + Ci * qr;
    s_ar[n] = ar;
    s_ai[n] = ai;
    float er = expf(64.0f * ar);                           // w^T, closed form
    wT_re[h * N2 + n] = er * cosf(64.0f * ai);
    wT_im[h * N2 + n] = er * sinf(64.0f * ai);
  }
  __syncthreads();
  {
    float d = (float)t;                                    // k[d] = 2Re sum C'w^d
    float acc = 0.f;
    for (int n = 0; n < N2; ++n) {
      float er = expf(d * s_ar[n]);
      float pr = er * cosf(d * s_ai[n]);
      float pi = er * sinf(d * s_ai[n]);
      acc += s_cr[n] * pr - s_ci[n] * pi;
    }
    s_k[t] = 2.0f * acc;
  }
  __syncthreads();
  size_t base = (size_t)h * (T_CHUNK * T_CHUNK);
  for (int j = 0; j < T_CHUNK; ++j)
    Ktoep[base + t * T_CHUNK + j] = f2bf(j <= t ? s_k[t - j] : 0.0f);
  for (int n = 0; n < N2; ++n) {
    float d = (float)(t + 1);          // P row l=t: [2Re(C'w^{l+1}), -2Im(...)]
    float er = expf(d * s_ar[n]);
    float pr = er * cosf(d * s_ai[n]), pi = er * sinf(d * s_ai[n]);
    float gr = s_cr[n] * pr - s_ci[n] * pi;
    float gi = s_cr[n] * pi + s_ci[n] * pr;
    Ptab[base + t * T_CHUNK + 2 * n]     = f2bf(2.0f * gr);
    Ptab[base + t * T_CHUNK + 2 * n + 1] = f2bf(-2.0f * gi);
    float dq = (float)(T_CHUNK - 1 - t);                   // Q col j=t: w^{T-1-j}
    float eq = expf(dq * s_ar[n]);
    Qtab[base + (2 * n) * T_CHUNK + t]     = f2bf(eq * cosf(dq * s_ai[n]));
    Qtab[base + (2 * n + 1) * T_CHUNK + t] = f2bf(eq * sinf(dq * s_ai[n]));
  }
}

// ---------------------------------------------------------------------------
// 2) Encoder GEMM (2x4 blocked, B staged in LDS): u = x @ enc_w^T + enc_b.
//    Block: 4 waves, same n-group, consecutive m-groups. KS=4.
// ---------------------------------------------------------------------------
__global__ void encoder_kernel(const __bf16* __restrict__ xpack,
                               const __bf16* __restrict__ wpack,
                               const float* __restrict__ enc_b,
                               float* __restrict__ u_ch) {
  __shared__ uint4 bsh4[16 * 64];            // 4 tiles x 4 ks x 1KB = 16KB
  int t = threadIdx.x;
  int mgs = blockIdx.x & 511;                // 512 m-superblocks
  int ng = blockIdx.x >> 9;                  // 0..3
  {
    const uint4* src = (const uint4*)wpack + (size_t)ng * 16 * 64;
#pragma unroll
    for (int i = 0; i < 8; ++i) bsh4[i * 128 + t] = src[i * 128 + t];
  }
  __syncthreads();

  int lane = t & 31;
  int wv = t >> 5;
  int mg = mgs * 4 + wv;                     // m-group of 2 tiles (0..2047)
  int am = lane & 15, half = lane >> 4;
  const v16bf* Ap = (const v16bf*)xpack;
  const v16bf* Bs = (const v16bf*)bsh4;
  v8f acc[2][4] = {};
#pragma unroll
  for (int ks = 0; ks < 4; ++ks) {
    v16bf a[2];
#pragma unroll
    for (int i = 0; i < 2; ++i)
      a[i] = Ap[(((size_t)(mg * 2 + i)) * 4 + ks) * 32 + lane];
#pragma unroll
    for (int j = 0; j < 4; ++j) {
      v16bf b = Bs[(j * 4 + ks) * 32 + lane];
#pragma unroll
      for (int i = 0; i < 2; ++i) acc[i][j] = wmma_bf16(a[i], b, acc[i][j]);
    }
  }
#pragma unroll
  for (int i = 0; i < 2; ++i) {
#pragma unroll
    for (int j = 0; j < 4; ++j) {
      int hch = (ng * 4 + j) * 16 + am;
      float bias = enc_b[hch];
#pragma unroll
      for (int r = 0; r < 8; ++r) {
        int m = (mg * 2 + i) * 16 + r + 8 * half;
        int bb = m >> 12, l = m & 4095;
        u_ch[((size_t)(bb * D_MODEL + hch)) * L_SEQ + l] = acc[i][j][r] + bias;
      }
    }
  }
}

// ---------------------------------------------------------------------------
// 3) Chunk-local end states: S_loc[2N x 16b] = Q[2N x T] @ U[T x 16b].
//    One block per (h,c); u-chunk staged once in LDS, shared by 4 waves.
// ---------------------------------------------------------------------------
__global__ void chunkstate_kernel(const float* __restrict__ u_ch,
                                  const __bf16* __restrict__ Qtab,
                                  float* __restrict__ s_loc) {
  __shared__ float ufl[B_SZ][LDS_PITCH];       // [b][j]
  int t = threadIdx.x;                         // 128 threads
  int h = blockIdx.x >> 6;
  int c = blockIdx.x & 63;
#pragma unroll
  for (int i = 0; i < 8; ++i) {
    int idx = i * 128 + t;                     // 1024 elements
    int b = idx >> 6, j = idx & 63;
    ufl[b][j] = u_ch[((size_t)(b * D_MODEL + h)) * L_SEQ + c * T_CHUNK + j];
  }
  __syncthreads();

  int lane = t & 31;
  int mt = t >> 5;                             // state tile 0..3
  int am = lane & 15, half = lane >> 4;
  const __bf16* Qrow =
      Qtab + (size_t)h * (T_CHUNK * T_CHUNK) + (mt * 16 + am) * T_CHUNK + 8 * half;
  v8f acc = {};
#pragma unroll
  for (int ks = 0; ks < 2; ++ks) {
    int kk = ks * 32;
    v16bf a, b;
#pragma unroll
    for (int i = 0; i < 8; ++i) {              // two contiguous 8-runs
      a[i]     = Qrow[kk + i];
      a[8 + i] = Qrow[kk + 16 + i];
    }
    const float* ub = &ufl[am][kk + 8 * half];
#pragma unroll
    for (int e = 0; e < 16; ++e) b[e] = f2bf(ub[e + (e >= 8 ? 8 : 0)]);
    acc = wmma_bf16(a, b, acc);
  }
  size_t base = ((size_t)(h * N_CHUNK + c)) * (2 * N2) * 16;
#pragma unroll
  for (int r = 0; r < 8; ++r) {
    int s = mt * 16 + r + 8 * half;
    s_loc[base + s * 16 + am] = acc[r];
  }
}

// ---------------------------------------------------------------------------
// 4) Serial scan over 64 chunks (per b,h,n): s0[c+1] = w^T s0[c] + s_loc[c].
//    In-place: s_loc[c] is replaced by the chunk's *initial* state s0[c].
// ---------------------------------------------------------------------------
__global__ void chunkscan_kernel(float* __restrict__ s_loc,
                                 const float* __restrict__ wT_re,
                                 const float* __restrict__ wT_im) {
  int tid = blockIdx.x * blockDim.x + threadIdx.x;  // 131072
  int bb = tid & 15;
  int n = (tid >> 4) & 31;
  int h = tid >> 9;
  float wr = wT_re[h * N2 + n], wi = wT_im[h * N2 + n];
  float sr = 0.f, si = 0.f;
  for (int c = 0; c < N_CHUNK; ++c) {
    size_t base = ((size_t)(h * N_CHUNK + c)) * (2 * N2) * 16;
    size_t ir = base + (2 * n) * 16 + bb;
    size_t ii = base + (2 * n + 1) * 16 + bb;
    float lr = s_loc[ir], li = s_loc[ii];
    s_loc[ir] = sr;
    s_loc[ii] = si;
    float nr = wr * sr - wi * si + lr;
    float ni = wr * si + wi * sr + li;
    sr = nr;
    si = ni;
  }
}

// ---------------------------------------------------------------------------
// 5) Conv output: Y = Ktoep @ U + P @ s0, then Dskip, GELU, residual.
//    One block per (h,c); u-chunk and s0 staged once in LDS.
// ---------------------------------------------------------------------------
__global__ void chunkconv_kernel(const float* __restrict__ u_ch,
                                 const __bf16* __restrict__ Ktoep,
                                 const __bf16* __restrict__ Ptab,
                                 const float* __restrict__ s0,
                                 const float* __restrict__ Dskip,
                                 __bf16* __restrict__ hpost) {
  __shared__ float ufl[B_SZ][LDS_PITCH];       // [b][j]
  __shared__ float s0s[B_SZ][LDS_PITCH];       // [b][s]
  int t = threadIdx.x;                         // 128 threads
  int h = blockIdx.x >> 6;
  int c = blockIdx.x & 63;
  size_t sbase = ((size_t)(h * N_CHUNK + c)) * (2 * N2) * 16;
#pragma unroll
  for (int i = 0; i < 8; ++i) {
    int idx = i * 128 + t;                     // u: consecutive lanes -> j
    int b = idx >> 6, j = idx & 63;
    ufl[b][j] = u_ch[((size_t)(b * D_MODEL + h)) * L_SEQ + c * T_CHUNK + j];
  }
#pragma unroll
  for (int i = 0; i < 8; ++i) {
    int idx = i * 128 + t;                     // s0: fully coalesced, transpose
    int s = idx >> 4, b = idx & 15;
    s0s[b][s] = s0[sbase + idx];
  }
  __syncthreads();

  int lane = t & 31;
  int lt = t >> 5;                             // l tile 0..3 (wave-uniform)
  int am = lane & 15, half = lane >> 4;
  const __bf16* Krow =
      Ktoep + (size_t)h * (T_CHUNK * T_CHUNK) + (lt * 16 + am) * T_CHUNK + 8 * half;
  const __bf16* Prow =
      Ptab + (size_t)h * (T_CHUNK * T_CHUNK) + (lt * 16 + am) * T_CHUNK + 8 * half;
  v8f acc = {};
  // intra-chunk causal Toeplitz (skip all-zero K blocks; wave-uniform branch)
  int ksteps = (lt >= 2) ? 2 : 1;
  for (int ks = 0; ks < ksteps; ++ks) {
    int kk = ks * 32;
    v16bf a, b;
#pragma unroll
    for (int i = 0; i < 8; ++i) {
      a[i]     = Krow[kk + i];
      a[8 + i] = Krow[kk + 16 + i];
    }
    const float* ub = &ufl[am][kk + 8 * half];
#pragma unroll
    for (int e = 0; e < 16; ++e) b[e] = f2bf(ub[e + (e >= 8 ? 8 : 0)]);
    acc = wmma_bf16(a, b, acc);
  }
  // initial-state contribution: P[T x 2N] @ s0[2N x 16b]
#pragma unroll
  for (int ks = 0; ks < 2; ++ks) {
    int kk = ks * 32;
    v16bf a, b;
#pragma unroll
    for (int i = 0; i < 8; ++i) {
      a[i]     = Prow[kk + i];
      a[8 + i] = Prow[kk + 16 + i];
    }
    const float* sb = &s0s[am][kk + 8 * half];
#pragma unroll
    for (int e = 0; e < 16; ++e) b[e] = f2bf(sb[e + (e >= 8 ? 8 : 0)]);
    acc = wmma_bf16(a, b, acc);
  }
  float dsk = Dskip[h];
#pragma unroll
  for (int r = 0; r < 8; ++r) {
    int ml = r + 8 * half;
    int lloc = lt * 16 + ml;
    int l = c * T_CHUNK + lloc;
    float uval = ufl[am][lloc];
    float y = acc[r] + uval * dsk;
    float hv = gelu_f(y) + uval;
    hpost[((size_t)(am * L_SEQ + l)) * D_MODEL + h] = f2bf(hv);
  }
}

// ---------------------------------------------------------------------------
// 6) LayerNorm over channels: one wave per (b,l) row, shfl_xor reduction.
// ---------------------------------------------------------------------------
__global__ void layernorm_kernel(const __bf16* __restrict__ hpost,
                                 const float* __restrict__ ln_g,
                                 const float* __restrict__ ln_b,
                                 __bf16* __restrict__ tout) {
  int lane = threadIdx.x & 31;
  int wv = threadIdx.x >> 5;
  int row = blockIdx.x * 4 + wv;    // 0..65535
  const __bf16* hp = hpost + (size_t)row * D_MODEL;
  float vals[8];
  float s = 0.f, s2 = 0.f;
#pragma unroll
  for (int i = 0; i < 8; ++i) {
    float v = bf2f(hp[i * 32 + lane]);
    vals[i] = v;
    s += v;
    s2 += v * v;
  }
#pragma unroll
  for (int o = 16; o > 0; o >>= 1) {
    s += __shfl_xor(s, o, 32);
    s2 += __shfl_xor(s2, o, 32);
  }
  float mu = s * (1.0f / D_MODEL);
  float var = s2 * (1.0f / D_MODEL) - mu * mu;
  float rstd = rsqrtf(var + 1e-5f);
  __bf16* tp = tout + (size_t)row * D_MODEL;
#pragma unroll
  for (int i = 0; i < 8; ++i) {
    int hch = i * 32 + lane;
    tp[hch] = f2bf((vals[i] - mu) * rstd * ln_g[hch] + ln_b[hch]);
  }
}

// ---------------------------------------------------------------------------
// 7) Decoder layer 1 (2x4 blocked, B staged in LDS):
//    mid = gelu(t @ dec1_w^T + b1). KS=8.
// ---------------------------------------------------------------------------
__global__ void dec1_kernel(const __bf16* __restrict__ tin,
                            const __bf16* __restrict__ w1pack,
                            const float* __restrict__ b1,
                            __bf16* __restrict__ mid) {
  __shared__ uint4 bsh4[32 * 64];            // 4 tiles x 8 ks x 1KB = 32KB
  int t = threadIdx.x;
  int mgs = blockIdx.x & 511;                // 512 m-superblocks
  int ng = blockIdx.x >> 9;                  // 0..7
  {
    const uint4* src = (const uint4*)w1pack + (size_t)ng * 32 * 64;
#pragma unroll
    for (int i = 0; i < 16; ++i) bsh4[i * 128 + t] = src[i * 128 + t];
  }
  __syncthreads();

  int lane = t & 31;
  int wv = t >> 5;
  int mg = mgs * 4 + wv;                     // m-group of 2 tiles (0..2047)
  int am = lane & 15, half = lane >> 4;
  const __bf16* ap[2];
#pragma unroll
  for (int i = 0; i < 2; ++i)
    ap[i] = tin + (size_t)(mg * 32 + i * 16 + am) * D_MODEL + 8 * half;
  const v16bf* Bs = (const v16bf*)bsh4;
  v8f acc[2][4] = {};
#pragma unroll 1
  for (int ks = 0; ks < 8; ++ks) {
    int kk = ks * 32;
    v16bf a[2];
#pragma unroll
    for (int i = 0; i < 2; ++i) {
      __builtin_prefetch(ap[i] + kk + 64, 0, 3);
#pragma unroll
      for (int e = 0; e < 8; ++e) {
        a[i][e]     = ap[i][kk + e];
        a[i][8 + e] = ap[i][kk + 16 + e];
      }
    }
#pragma unroll
    for (int j = 0; j < 4; ++j) {
      v16bf b = Bs[(j * 8 + ks) * 32 + lane];
#pragma unroll
      for (int i = 0; i < 2; ++i) acc[i][j] = wmma_bf16(a[i], b, acc[i][j]);
    }
  }
#pragma unroll
  for (int i = 0; i < 2; ++i) {
#pragma unroll
    for (int j = 0; j < 4; ++j) {
      int n = (ng * 4 + j) * 16 + am;
      float bias = b1[n];
#pragma unroll
      for (int r = 0; r < 8; ++r) {
        int m = mg * 32 + i * 16 + r + 8 * half;
        mid[(size_t)m * D_HID + n] = f2bf(gelu_f(acc[i][j][r] + bias));
      }
    }
  }
}

// ---------------------------------------------------------------------------
// 8) Decoder layer 2 (2x4 blocked, B staged in LDS):
//    out = mid @ dec2_w^T + b2.  KS=16; 7 n-tiles split as groups of 4+3.
// ---------------------------------------------------------------------------
__global__ void dec2_kernel(const __bf16* __restrict__ mid,
                            const __bf16* __restrict__ w2pack,
                            const float* __restrict__ b2,
                            float* __restrict__ out) {
  __shared__ uint4 bsh4[64 * 64];            // up to 4 tiles x 16 ks x 1KB = 64KB
  int t = threadIdx.x;
  int mgs = blockIdx.x & 511;                // 512 m-superblocks
  int ng = blockIdx.x >> 9;                  // 0..1 (tiles 0-3 / 4-6)
  int nfr = (ng == 0) ? 64 : 48;             // fragments to stage
  {
    const uint4* src = (const uint4*)w2pack + (size_t)ng * 64 * 64;
    int n4 = nfr * 64;
    for (int idx = t; idx < n4; idx += 128) bsh4[idx] = src[idx];
  }
  __syncthreads();

  int lane = t & 31;
  int wv = t >> 5;
  int mg = mgs * 4 + wv;                     // m-group of 2 tiles (0..2047)
  int am = lane & 15, half = lane >> 4;
  const __bf16* ap[2];
#pragma unroll
  for (int i = 0; i < 2; ++i)
    ap[i] = mid + (size_t)(mg * 32 + i * 16 + am) * D_HID + 8 * half;
  const v16bf* Bs = (const v16bf*)bsh4;
  v8f acc[2][4] = {};
#pragma unroll 1
  for (int ks = 0; ks < 16; ++ks) {
    int kk = ks * 32;
    v16bf a[2];
#pragma unroll
    for (int i = 0; i < 2; ++i) {
      __builtin_prefetch(ap[i] + kk + 64, 0, 3);
#pragma unroll
      for (int e = 0; e < 8; ++e) {
        a[i][e]     = ap[i][kk + e];
        a[i][8 + e] = ap[i][kk + 16 + e];
      }
    }
#pragma unroll
    for (int j = 0; j < 4; ++j) {            // j=3 of ng=1 is padding (discarded)
      v16bf b = Bs[(j * 16 + ks) * 32 + lane];
#pragma unroll
      for (int i = 0; i < 2; ++i) acc[i][j] = wmma_bf16(a[i], b, acc[i][j]);
    }
  }
#pragma unroll
  for (int i = 0; i < 2; ++i) {
#pragma unroll
    for (int j = 0; j < 4; ++j) {
      int tile = ng * 4 + j;
      int col = tile * 16 + am;
      if (tile < 7 && col < D_OUTPUT) {
        float bias = b2[col];
#pragma unroll
        for (int r = 0; r < 8; ++r) {
          int m = mg * 32 + i * 16 + r + 8 * half;
          out[(size_t)m * D_OUTPUT + col] = acc[i][j][r] + bias;
        }
      }
    }
  }
}

// ---------------------------------------------------------------------------
extern "C" void kernel_launch(void* const* d_in, const int* in_sizes, int n_in,
                              void* d_out, int out_size, void* d_ws, size_t ws_size,
                              hipStream_t stream) {
  (void)in_sizes; (void)n_in; (void)out_size; (void)ws_size;
  const float* x      = (const float*)d_in[0];
  const float* enc_w  = (const float*)d_in[1];
  const float* enc_b  = (const float*)d_in[2];
  const float* log_dt = (const float*)d_in[3];
  const float* log_A  = (const float*)d_in[4];
  const float* A_im   = (const float*)d_in[5];
  const float* B_re   = (const float*)d_in[6];
  const float* B_im   = (const float*)d_in[7];
  const float* C_re   = (const float*)d_in[8];
  const float* C_im   = (const float*)d_in[9];
  const float* Dskip  = (const float*)d_in[10];
  const float* ln_g   = (const float*)d_in[11];
  const float* ln_b   = (const float*)d_in[12];
  const float* w1     = (const float*)d_in[13];
  const float* b1     = (const float*)d_in[14];
  const float* w2     = (const float*)d_in[15];
  const float* b2     = (const float*)d_in[16];
  float* out = (float*)d_out;

  char* ws = (char*)d_ws;
  size_t off = 0;
  auto take = [&](size_t bytes) {
    char* p = ws + off;
    off += (bytes + 255) & ~(size_t)255;
    return p;
  };
  float*  u_ch  = (float*)take((size_t)B_SZ * D_MODEL * L_SEQ * 4);          // 64 MB
  float*  s_loc = (float*)take((size_t)D_MODEL * N_CHUNK * 2 * N2 * 16 * 4); // 64 MB
  __bf16* hpost = (__bf16*)take((size_t)B_SZ * L_SEQ * D_MODEL * 2);         // 32 MB
  __bf16* tbuf  = (__bf16*)take((size_t)B_SZ * L_SEQ * D_MODEL * 2);         // 32 MB
  __bf16* mid   = (__bf16*)take((size_t)B_SZ * L_SEQ * D_HID * 2);           // 64 MB
  __bf16* Ktoep = (__bf16*)take((size_t)D_MODEL * T_CHUNK * T_CHUNK * 2);
  __bf16* Ptab  = (__bf16*)take((size_t)D_MODEL * T_CHUNK * T_CHUNK * 2);
  __bf16* Qtab  = (__bf16*)take((size_t)D_MODEL * T_CHUNK * T_CHUNK * 2);
  float*  wT_re = (float*)take((size_t)D_MODEL * N2 * 4);
  float*  wT_im = (float*)take((size_t)D_MODEL * N2 * 4);
  __bf16* xpack = (__bf16*)take((size_t)4096 * 4 * 32 * 16 * 2);             // 16.8 MB
  __bf16* wencp = (__bf16*)take((size_t)16 * 4 * 32 * 16 * 2);
  __bf16* w1p   = (__bf16*)take((size_t)32 * 8 * 32 * 16 * 2);
  __bf16* w2p   = (__bf16*)take((size_t)7 * 16 * 32 * 16 * 2);

  // packing passes (tiny except xpack; all coalesced writes)
  pack_a_x_kernel<<<2048, 256, 0, stream>>>(x, xpack);
  pack_b_kernel<<<8, 256, 0, stream>>>(enc_w, wencp, D_MODEL, D_INPUT, 4);
  pack_b_kernel<<<32, 256, 0, stream>>>(w1, w1p, D_HID, D_MODEL, 8);
  pack_b_kernel<<<14, 256, 0, stream>>>(w2, w2p, D_OUTPUT, D_HID, 16);
  s4_tables_kernel<<<D_MODEL, T_CHUNK, 0, stream>>>(log_dt, log_A, A_im, B_re, B_im,
                                                    C_re, C_im, Ktoep, Ptab, Qtab,
                                                    wT_re, wT_im);
  encoder_kernel<<<2048, 128, 0, stream>>>(xpack, wencp, enc_b, u_ch);
  chunkstate_kernel<<<16384, 128, 0, stream>>>(u_ch, Qtab, s_loc);
  chunkscan_kernel<<<512, 256, 0, stream>>>(s_loc, wT_re, wT_im);
  chunkconv_kernel<<<16384, 128, 0, stream>>>(u_ch, Ktoep, Ptab, s_loc, Dskip, hpost);
  layernorm_kernel<<<16384, 128, 0, stream>>>(hpost, ln_g, ln_b, tbuf);
  dec1_kernel<<<4096, 128, 0, stream>>>(tbuf, w1p, b1, mid);
  dec2_kernel<<<1024, 128, 0, stream>>>(mid, w2p, b2, out);
}
